// MultiHeadAttentionScore_36782099923461
// MI455X (gfx1250) — matrix operand order repaired and checked by reading, not compiled, not run on hardware
//
#include <hip/hip_runtime.h>
#include <hip/hip_bf16.h>

// ---------------- constants matching the reference ----------------
constexpr int C_B = 8, C_S = 512, C_DIN = 256, C_DOUT = 256;
constexpr int C_H = 8, C_L = 4, C_R = 8, C_DEPTH = 32;
constexpr int C_N = C_B * C_S;                 // 4096 nodes

// ---------------- WMMA vector types ----------------
typedef __attribute__((ext_vector_type(16))) __bf16 v16bf;
typedef __attribute__((ext_vector_type(8)))  float  v8f;

struct Frag { union { v16bf v; unsigned int w[8]; }; };

// float -> bf16 (round to nearest even)
static __device__ __forceinline__ unsigned short f2bf(float f) {
  unsigned u = __float_as_uint(f);
  u = u + 0x7FFFu + ((u >> 16) & 1u);
  return (unsigned short)(u >> 16);
}

// order-preserving float<->uint encoding for atomic max
static __device__ __forceinline__ unsigned fenc(float f) {
  unsigned u = __float_as_uint(f);
  return (u & 0x80000000u) ? ~u : (u | 0x80000000u);
}
static __device__ __forceinline__ float fdec(unsigned u) {
  unsigned v = (u & 0x80000000u) ? (u & 0x7FFFFFFFu) : ~u;
  return __uint_as_float(v);
}

// A-matrix 16x32 bf16 fragment (ISA 7.12.2): lanes 0-15 row M, K={0..7,16..23};
// lanes 16-31 same rows, K={8..15,24..31}.  Two 16B contiguous chunks.
static __device__ __forceinline__ v16bf load_a_frag(const unsigned short* __restrict__ base,
                                                    int ld, int row, int half) {
  Frag f;
  const unsigned short* p = base + (long)row * ld + 8 * half;
#pragma unroll
  for (int j = 0; j < 4; ++j) f.w[j] = *(const unsigned int*)(p + 2 * j);
#pragma unroll
  for (int j = 0; j < 4; ++j) f.w[4 + j] = *(const unsigned int*)(p + 16 + 2 * j);
  return f.v;
}

// B-matrix 32x16 bf16 fragment: lanes 0-15 hold K={0..7,16..23}, lanes 16-31 K={8..15,24..31}
// (two 16B chunks), column N = lane&15.  Stored as Bt[N,K] row-major.
static __device__ __forceinline__ v16bf load_b_frag(const unsigned short* __restrict__ base,
                                                    int ld, int row, int half) {
  Frag f;
  const unsigned short* p = base + (long)row * ld + 8 * half;
#pragma unroll
  for (int j = 0; j < 8; ++j) f.w[j] = *(const unsigned int*)(p + (j < 4 ? 2 * j : 16 + 2 * (j - 4)));
  return f.v;
}

static __device__ __forceinline__ v8f wmma_bf16(v16bf a, v16bf b, v8f c) {
  return __builtin_amdgcn_wmma_f32_16x16x32_bf16(false, a, false, b, (short)0, c, false, false);
}

// ---- generic bf16 WMMA GEMM, 32x32 register tile per wave ----
// C[M,N] = A[M,K] * Bt[N,K]^T + bias.  blockDim=(32,4); grid=(N/32, M/128, batch)
__global__ void gemm_bf16_wmma(const unsigned short* __restrict__ A,
                               const unsigned short* __restrict__ Bt,
                               const float* __restrict__ bias,
                               float* __restrict__ C,
                               int M, int Nn, int K,
                               long aBatch, long bBatch, long cBatch) {
  const int lane = threadIdx.x;
  const int half = lane >> 4, idx = lane & 15;
  const int tN = blockIdx.x;
  const int tM = blockIdx.y * blockDim.y + threadIdx.y;
  const int bat = blockIdx.z;
  const unsigned short* Ab = A + (long)bat * aBatch;
  const unsigned short* Bb = Bt + (long)bat * bBatch;
  float* Cb = C + (long)bat * cBatch;
  const int m0 = tM * 32, n0 = tN * 32;

  v8f acc00 = {}, acc01 = {}, acc10 = {}, acc11 = {};
  for (int k0 = 0; k0 < K; k0 += 32) {
    v16bf a0 = load_a_frag(Ab + k0, K, m0 + idx, half);
    v16bf a1 = load_a_frag(Ab + k0, K, m0 + 16 + idx, half);
    v16bf b0 = load_b_frag(Bb + k0, K, n0 + idx, half);
    v16bf b1 = load_b_frag(Bb + k0, K, n0 + 16 + idx, half);
    acc00 = wmma_bf16(a0, b0, acc00);
    acc01 = wmma_bf16(a0, b1, acc01);
    acc10 = wmma_bf16(a1, b0, acc10);
    acc11 = wmma_bf16(a1, b1, acc11);
  }
  const float bv0 = bias ? bias[n0 + idx] : 0.0f;
  const float bv1 = bias ? bias[n0 + 16 + idx] : 0.0f;
#pragma unroll
  for (int i = 0; i < 8; ++i) {
    const long r0 = (long)(m0 + i + 8 * half) * Nn;
    const long r1 = (long)(m0 + 16 + i + 8 * half) * Nn;
    Cb[r0 + n0 + idx]      = acc00[i] + bv0;
    Cb[r0 + n0 + 16 + idx] = acc01[i] + bv1;
    Cb[r1 + n0 + idx]      = acc10[i] + bv0;
    Cb[r1 + n0 + 16 + idx] = acc11[i] + bv1;
  }
}

// ---- attention scores: out[b,h,q,k] = (q . k) * scale, 32x32 tile per wave ----
// DEPTH = 32 -> one k-step, 4 WMMAs.  blockDim=(32,4); grid=(S/32, S/128, B*H)
__global__ void attn_wmma(const unsigned short* __restrict__ Q,
                          const unsigned short* __restrict__ Kt,
                          float* __restrict__ out, float scale) {
  const int lane = threadIdx.x;
  const int half = lane >> 4, idx = lane & 15;
  const int tK = blockIdx.x;
  const int tQ = blockIdx.y * blockDim.y + threadIdx.y;
  const int bh = blockIdx.z;
  const int b = bh / C_H, h = bh % C_H;
  const unsigned short* qb = Q + (long)b * C_S * C_DOUT + h * C_DEPTH;
  const unsigned short* kb = Kt + (long)b * C_S * C_DOUT + h * C_DEPTH;
  const int q0 = tQ * 32, k0 = tK * 32;

  v16bf a0 = load_a_frag(qb, C_DOUT, q0 + idx, half);
  v16bf a1 = load_a_frag(qb, C_DOUT, q0 + 16 + idx, half);
  v16bf b0 = load_b_frag(kb, C_DOUT, k0 + idx, half);
  v16bf b1 = load_b_frag(kb, C_DOUT, k0 + 16 + idx, half);
  v8f acc00 = {}, acc01 = {}, acc10 = {}, acc11 = {};
  acc00 = wmma_bf16(a0, b0, acc00);
  acc01 = wmma_bf16(a0, b1, acc01);
  acc10 = wmma_bf16(a1, b0, acc10);
  acc11 = wmma_bf16(a1, b1, acc11);

  float* ob = out + (long)bh * C_S * C_S;
#pragma unroll
  for (int i = 0; i < 8; ++i) {
    const long r0 = (long)(q0 + i + 8 * half) * C_S;
    const long r1 = (long)(q0 + 16 + i + 8 * half) * C_S;
    ob[r0 + k0 + idx]      = acc00[i] * scale;
    ob[r0 + k0 + 16 + idx] = acc01[i] * scale;
    ob[r1 + k0 + idx]      = acc10[i] * scale;
    ob[r1 + k0 + 16 + idx] = acc11[i] * scale;
  }
}

// ---------------- converters ----------------
__global__ void conv_f32_bf16(const float* __restrict__ in, unsigned short* __restrict__ out, int n) {
  int i = blockIdx.x * blockDim.x + threadIdx.x;
  if (i < n) out[i] = f2bf(in[i]);
}
// out[b][c*rows + r] = in[b][r*cols + c]   (transpose + convert, per batch)
__global__ void conv_t_bf16(const float* __restrict__ in, unsigned short* __restrict__ out,
                            int rows, int cols, int total) {
  int i = blockIdx.x * blockDim.x + threadIdx.x;
  if (i >= total) return;
  int per = rows * cols;
  int b = i / per, rem = i - b * per;
  int r = rem / cols, c = rem - r * cols;
  out[(long)b * per + (long)c * rows + r] = f2bf(in[i]);
}

__global__ void fill_u32(unsigned* __restrict__ p, unsigned v, int n) {
  int i = blockIdx.x * blockDim.x + threadIdx.x;
  if (i < n) p[i] = v;
}

// ---------------- per-(relation,node) dot products ----------------
// srcdot[r*N+n] = a_src[l,r] . hr[r,n,:]   dstdot[r*N+n] = a_dst[l,r] . h[n,:]
// blockDim=(32,8); grid = R*N/8
__global__ void rel_dots(const float* __restrict__ hr, const float* __restrict__ h,
                         const float* __restrict__ a_src, const float* __restrict__ a_dst,
                         float* __restrict__ srcdot, float* __restrict__ dstdot, int l) {
  const int lane = threadIdx.x;
  const int rn = blockIdx.x * blockDim.y + threadIdx.y;
  const int r = rn >> 12;            // N = 4096
  const int n = rn & (C_N - 1);
  const float* as = a_src + ((long)l * C_R + r) * C_DOUT;
  const float* ad = a_dst + ((long)l * C_R + r) * C_DOUT;
  const float* hv = hr + (long)rn * C_DOUT;     // rn == r*N + n
  const float* hn = h + (long)n * C_DOUT;
  float s1 = 0.f, s2 = 0.f;
  for (int t = lane; t < C_DOUT; t += 32) { s1 += as[t] * hv[t]; s2 += ad[t] * hn[t]; }
#pragma unroll
  for (int off = 16; off; off >>= 1) {
    s1 += __shfl_xor(s1, off, 32);
    s2 += __shfl_xor(s2, off, 32);
  }
  if (lane == 0) { srcdot[rn] = s1; dstdot[rn] = s2; }
}

// ---------------- edge kernels ----------------
__global__ void edge_logits_max(const int* __restrict__ etype, const int* __restrict__ esrc,
                                const int* __restrict__ edst,
                                const float* __restrict__ srcdot, const float* __restrict__ dstdot,
                                float* __restrict__ lg, unsigned* __restrict__ nodemax, int E) {
  int e = blockIdx.x * blockDim.x + threadIdx.x;
  if (e >= E) return;
  int r = etype[e], s = esrc[e], d = edst[e];
  float lo = srcdot[r * C_N + s] + dstdot[r * C_N + d];
  float lk = lo > 0.f ? lo : 0.2f * lo;        // leaky_relu(0.2)
  lg[e] = lk;
  atomicMax(nodemax + d, fenc(lk));
}

__global__ void edge_exp_sum(const int* __restrict__ edst,
                             const float* __restrict__ lg, const unsigned* __restrict__ nodemax,
                             float* __restrict__ ex, float* __restrict__ denom, int E) {
  int e = blockIdx.x * blockDim.x + threadIdx.x;
  if (e >= E) return;
  int d = edst[e];
  float m = fdec(nodemax[d]);
  float x = __expf(lg[e] - m);
  ex[e] = x;
  atomicAdd(denom + d, x);
}

// one thread per (edge, 4 features): 64 threads/edge
__global__ void edge_aggregate(const int* __restrict__ etype, const int* __restrict__ esrc,
                               const int* __restrict__ edst,
                               const float* __restrict__ ex, const float* __restrict__ denom,
                               const float* __restrict__ hr, float* __restrict__ agg, int E) {
  int t = blockIdx.x * blockDim.x + threadIdx.x;
  if (t >= E * (C_DOUT / 4)) return;
  int e = t >> 6;
  int j = (t & 63) * 4;
  int r = etype[e], s = esrc[e], d = edst[e];
  float alpha = ex[e] / (denom[d] + 1e-16f);
  const float4 m = *(const float4*)(hr + ((long)r * C_N + s) * C_DOUT + j);
  float* ap = agg + (long)d * C_DOUT + j;
  atomicAdd(ap + 0, alpha * m.x);
  atomicAdd(ap + 1, alpha * m.y);
  atomicAdd(ap + 2, alpha * m.z);
  atomicAdd(ap + 3, alpha * m.w);
}

__global__ void elu_convert(const float* __restrict__ agg, float* __restrict__ h,
                            unsigned short* __restrict__ h_bf, int n) {
  int i = blockIdx.x * blockDim.x + threadIdx.x;
  if (i >= n) return;
  float v = agg[i];
  float o = v > 0.f ? v : (__expf(v) - 1.f);
  h[i] = o;
  h_bf[i] = f2bf(o);
}

// =============================== host side ===============================
extern "C" void kernel_launch(void* const* d_in, const int* in_sizes, int n_in,
                              void* d_out, int out_size, void* d_ws, size_t ws_size,
                              hipStream_t stream) {
  const float* x     = (const float*)d_in[0];
  const int*   esrc  = (const int*)d_in[1];
  const int*   edst  = (const int*)d_in[2];
  const int*   etype = (const int*)d_in[3];
  const float* Wl    = (const float*)d_in[4];
  const float* bl    = (const float*)d_in[5];
  const float* Wgat  = (const float*)d_in[6];
  const float* a_src = (const float*)d_in[7];
  const float* a_dst = (const float*)d_in[8];
  const float* W1    = (const float*)d_in[9];
  const float* b1    = (const float*)d_in[10];
  const float* Wq    = (const float*)d_in[11];
  const float* bq    = (const float*)d_in[12];
  const float* Wk    = (const float*)d_in[13];
  const float* bk    = (const float*)d_in[14];
  const int E = in_sizes[1];

  // ---- workspace carve-up ----
  char* w = (char*)d_ws;
  auto nxt = [&](size_t bytes) { char* p = w; w += (bytes + 255) & ~size_t(255); return p; };
  unsigned short* x_bf   = (unsigned short*)nxt((size_t)C_N * C_DIN * 2);
  unsigned short* Wl_bf  = (unsigned short*)nxt((size_t)C_DOUT * C_DIN * 2);
  unsigned short* W1_bf  = (unsigned short*)nxt((size_t)C_DIN * C_DOUT * 2);
  unsigned short* Wq_bf  = (unsigned short*)nxt((size_t)C_DOUT * C_DIN * 2);
  unsigned short* Wk_bf  = (unsigned short*)nxt((size_t)C_DOUT * C_DIN * 2);
  unsigned short* WgT_bf = (unsigned short*)nxt((size_t)C_L * C_R * C_DOUT * C_DOUT * 2);
  float*          h      = (float*)nxt((size_t)C_N * C_DOUT * 4);
  unsigned short* h_bf   = (unsigned short*)nxt((size_t)C_N * C_DOUT * 2);
  float*          hr     = (float*)nxt((size_t)C_R * C_N * C_DOUT * 4);   // 32 MB arena
  float*          srcdot = (float*)nxt((size_t)C_R * C_N * 4);
  float*          dstdot = (float*)nxt((size_t)C_R * C_N * 4);
  float*          lg     = (float*)nxt((size_t)E * 4);
  float*          ex     = (float*)nxt((size_t)E * 4);
  unsigned*       nmax   = (unsigned*)nxt((size_t)C_N * 4);
  float*          denom  = (float*)nxt((size_t)C_N * 4);
  float*          agg    = (float*)nxt((size_t)C_N * C_DOUT * 4);

  // q/k stage overlaid onto hr (dead after the GAT loop)
  float*          q      = hr;
  float*          k      = hr + (size_t)C_N * C_DOUT;           // +4MB
  float*          h1     = hr + 2 * (size_t)C_N * C_DOUT;       // +8MB
  unsigned short* h1_bf  = (unsigned short*)(hr + 3 * (size_t)C_N * C_DOUT);
  unsigned short* q_bf   = h1_bf + (size_t)C_N * C_DOUT;
  unsigned short* k_bf   = q_bf + (size_t)C_N * C_DOUT;

  const dim3 cvtB(256);
  auto cvtG = [](int n) { return dim3((n + 255) / 256); };

  // ---- weight / input conversion to bf16 ----
  conv_f32_bf16<<<cvtG(C_N * C_DIN), cvtB, 0, stream>>>(x, x_bf, C_N * C_DIN);
  conv_f32_bf16<<<cvtG(C_DOUT * C_DIN), cvtB, 0, stream>>>(Wl, Wl_bf, C_DOUT * C_DIN);
  conv_f32_bf16<<<cvtG(C_DIN * C_DOUT), cvtB, 0, stream>>>(W1, W1_bf, C_DIN * C_DOUT);
  conv_f32_bf16<<<cvtG(C_DOUT * C_DIN), cvtB, 0, stream>>>(Wq, Wq_bf, C_DOUT * C_DIN);
  conv_f32_bf16<<<cvtG(C_DOUT * C_DIN), cvtB, 0, stream>>>(Wk, Wk_bf, C_DOUT * C_DIN);
  {
    int total = C_L * C_R * C_DOUT * C_DOUT;
    conv_t_bf16<<<cvtG(total), cvtB, 0, stream>>>(Wgat, WgT_bf, C_DOUT, C_DOUT, total);
  }

  const dim3 gemmB(32, 4);
  // ---- h = x @ Wl^T + bl ----
  gemm_bf16_wmma<<<dim3(C_DOUT / 32, C_N / 128, 1), gemmB, 0, stream>>>(
      x_bf, Wl_bf, bl, h, C_N, C_DOUT, C_DIN, 0, 0, 0);
  conv_f32_bf16<<<cvtG(C_N * C_DOUT), cvtB, 0, stream>>>(h, h_bf, C_N * C_DOUT);

  // ---- 4 RGAT layers ----
  for (int l = 0; l < C_L; ++l) {
    // hr[r] = h @ WgatT[l,r]   (batched over r via grid.z)
    gemm_bf16_wmma<<<dim3(C_DOUT / 32, C_N / 128, C_R), gemmB, 0, stream>>>(
        h_bf, WgT_bf + (size_t)l * C_R * C_DOUT * C_DOUT, nullptr, hr,
        C_N, C_DOUT, C_DOUT,
        0, (long)C_DOUT * C_DOUT, (long)C_N * C_DOUT);

    rel_dots<<<dim3(C_R * C_N / 8), dim3(32, 8), 0, stream>>>(
        hr, h, a_src, a_dst, srcdot, dstdot, l);

    fill_u32<<<cvtG(C_N), cvtB, 0, stream>>>(nmax, 0u, C_N);
    fill_u32<<<cvtG(C_N), cvtB, 0, stream>>>((unsigned*)denom, 0u, C_N);
    fill_u32<<<cvtG(C_N * C_DOUT), cvtB, 0, stream>>>((unsigned*)agg, 0u, C_N * C_DOUT);

    edge_logits_max<<<cvtG(E), cvtB, 0, stream>>>(etype, esrc, edst, srcdot, dstdot, lg, nmax, E);
    edge_exp_sum<<<cvtG(E), cvtB, 0, stream>>>(edst, lg, nmax, ex, denom, E);
    edge_aggregate<<<cvtG(E * (C_DOUT / 4)), cvtB, 0, stream>>>(
        etype, esrc, edst, ex, denom, hr, agg, E);

    elu_convert<<<cvtG(C_N * C_DOUT), cvtB, 0, stream>>>(agg, h, h_bf, C_N * C_DOUT);
  }

  // ---- h1 = h @ W1^T + b1 ----
  gemm_bf16_wmma<<<dim3(C_DIN / 32, C_N / 128, 1), gemmB, 0, stream>>>(
      h_bf, W1_bf, b1, h1, C_N, C_DIN, C_DOUT, 0, 0, 0);
  conv_f32_bf16<<<cvtG(C_N * C_DIN), cvtB, 0, stream>>>(h1, h1_bf, C_N * C_DIN);

  // ---- q, k projections ----
  gemm_bf16_wmma<<<dim3(C_DOUT / 32, C_N / 128, 1), gemmB, 0, stream>>>(
      h1_bf, Wq_bf, bq, q, C_N, C_DOUT, C_DIN, 0, 0, 0);
  gemm_bf16_wmma<<<dim3(C_DOUT / 32, C_N / 128, 1), gemmB, 0, stream>>>(
      h1_bf, Wk_bf, bk, k, C_N, C_DOUT, C_DIN, 0, 0, 0);
  conv_f32_bf16<<<cvtG(C_N * C_DOUT), cvtB, 0, stream>>>(q, q_bf, C_N * C_DOUT);
  conv_f32_bf16<<<cvtG(C_N * C_DOUT), cvtB, 0, stream>>>(k, k_bf, C_N * C_DOUT);

  // ---- attention scores ----
  const float scale = 0.17677669529663687f;   // 1/sqrt(32)
  attn_wmma<<<dim3(C_S / 32, C_S / 128, C_B * C_H), gemmB, 0, stream>>>(
      q_bf, k_bf, (float*)d_out, scale);
}
